// CrossAttention_73650099191960
// MI455X (gfx1250) — compile-verified, matrix operand order checked by reading, not run on hardware
//
#include <hip/hip_runtime.h>
#include <hip/hip_bf16.h>
#include <math.h>

typedef __attribute__((ext_vector_type(16))) _Float16 v16h;
typedef __attribute__((ext_vector_type(8)))  _Float16 v8h;
typedef __attribute__((ext_vector_type(8)))  float    v8f;
typedef __attribute__((ext_vector_type(4)))  int      v4i;

#define NB   64      // B*S batches
#define CCH  512     // C1 == C2
#define HW   1024    // 32*32 pixels
#define LDSS 40      // LDS row stride in halfs (32 + 8 pad; 16B-aligned chunks, conflict-free)

// --------------------------------------------------------------------------
// CDNA5 async global->LDS path (ASYNCcnt-tracked, bypasses VGPRs).
// Builtin takes (v4i addrspace(1)* src, v4i addrspace(3)* dst, imm off, imm cpol).
// Falls back to plain 16B vector copies if the builtin is absent.
// --------------------------------------------------------------------------
#if defined(__has_builtin)
#if __has_builtin(__builtin_amdgcn_global_load_async_to_lds_b128)
#define USE_ASYNC_LDS 1
#endif
#endif

#if defined(USE_ASYNC_LDS)
typedef __attribute__((address_space(1))) v4i* gas_v4i;
typedef __attribute__((address_space(3))) v4i* las_v4i;
__device__ __forceinline__ void cp16_g2l(const void* g, void* l) {
  __builtin_amdgcn_global_load_async_to_lds_b128(
      (gas_v4i)(void*)g, (las_v4i)l, 0, 0);
}
__device__ __forceinline__ void cp16_wait() {
#if __has_builtin(__builtin_amdgcn_s_wait_asynccnt)
  __builtin_amdgcn_s_wait_asynccnt(0);
#else
  asm volatile("s_wait_asynccnt 0x0" ::: "memory");
#endif
}
#else
__device__ __forceinline__ void cp16_g2l(const void* g, void* l) {
  *(v8h*)l = *(const v8h*)g;
}
__device__ __forceinline__ void cp16_wait() {}
#endif

// --------------------------------------------------------------------------
// One 16x32 f16 WMMA A/B fragment from an LDS tile (row-major, stride LDSS).
// ISA 7.12.2 layout: lanes 0-15 -> row M=lane, K={0..7,16..23};
// lanes 16-31 -> row M=lane-16, K={8..15,24..31}. Two ds_load_b128 per lane.
// --------------------------------------------------------------------------
__device__ __forceinline__ v16h frag_ld(const _Float16* tile, int row0, int lane) {
  int r  = row0 + (lane & 15);
  int kb = (lane >> 4) << 3;
  const _Float16* p = tile + r * LDSS + kb;
  union { v16h v; v8h h[2]; } u;
  u.h[0] = *(const v8h*)(p);
  u.h[1] = *(const v8h*)(p + 16);
  return u.v;
}

// ===========================================================================
// Stage 1: Q/K/V = W * X + b   (M=512, N=1024, K=512), fp32 in -> f16 out.
// Block tile 128x256, 8 waves as 2(M)x4(N), wave tile 64x64 (16 WMMA/k-step).
// grid = (4*4 tiles, 3 which, 64 batch), block = 256.
// ===========================================================================
__global__ __launch_bounds__(256)
void gemm_qkv(const float* __restrict__ Wq, const float* __restrict__ bq,
              const float* __restrict__ Wk, const float* __restrict__ bk,
              const float* __restrict__ Wv, const float* __restrict__ bv,
              const float* __restrict__ img, const float* __restrict__ dep,
              _Float16* __restrict__ Q, _Float16* __restrict__ Kd,
              _Float16* __restrict__ V) {
  __shared__ _Float16 At[128 * LDSS];
  __shared__ _Float16 Bt[256 * LDSS];

  const int tid  = threadIdx.x;
  const int lane = tid & 31;
  const int wave = tid >> 5;
  const int wm   = (wave & 1) * 64;
  const int wn   = (wave >> 1) * 64;
  const int m0   = (blockIdx.x & 3) * 128;
  const int n0   = (blockIdx.x >> 2) * 256;
  const int b    = blockIdx.z;
  const int which = blockIdx.y;

  const float* Wm; const float* bias; const float* src; _Float16* dst;
  if (which == 0)      { Wm = Wq; bias = bq; src = img; dst = Q;  }
  else if (which == 1) { Wm = Wk; bias = bk; src = dep; dst = Kd; }
  else                 { Wm = Wv; bias = bv; src = dep; dst = V;  }
  const float* srcb = src + (size_t)b * CCH * HW;

  v8f acc[4][4];
  for (int i = 0; i < 4; ++i)
    for (int j = 0; j < 4; ++j) acc[i][j] = (v8f)(0.0f);

  for (int kt = 0; kt < CCH; kt += 32) {
    // A tile: W rows [m0,m0+128) x cols [kt,kt+32), cvt f32->f16
    for (int i = 0; i < 4; ++i) {
      int slot = tid + i * 256;                 // 1024 float4 slots
      int r    = slot >> 3;
      int c4   = (slot & 7) << 2;
      const float* gp = Wm + (size_t)(m0 + r) * CCH + kt + c4;
      float4 v = *(const float4*)gp;
      __builtin_prefetch(gp + 32, 0, 0);        // next k-tile -> global_prefetch
      _Float16* p = &At[r * LDSS + c4];
      p[0] = (_Float16)v.x; p[1] = (_Float16)v.y;
      p[2] = (_Float16)v.z; p[3] = (_Float16)v.w;
    }
    // B tile: X rows [kt,kt+32) x cols [n0,n0+256), transpose into [n][k]
    for (int i = 0; i < 8; ++i) {
      int slot = tid + i * 256;                 // 32 k-rows * 64 float4
      int kr   = slot >> 6;
      int n4   = (slot & 63) << 2;
      const float* gp = srcb + (size_t)(kt + kr) * HW + n0 + n4;
      float4 v = *(const float4*)gp;
      __builtin_prefetch(gp + 32 * HW, 0, 0);   // next k-tile
      Bt[(n4 + 0) * LDSS + kr] = (_Float16)v.x;
      Bt[(n4 + 1) * LDSS + kr] = (_Float16)v.y;
      Bt[(n4 + 2) * LDSS + kr] = (_Float16)v.z;
      Bt[(n4 + 3) * LDSS + kr] = (_Float16)v.w;
    }
    __syncthreads();

    v16h af[4], bf[4];
    for (int mi = 0; mi < 4; ++mi) af[mi] = frag_ld(At, wm + mi * 16, lane);
    for (int ni = 0; ni < 4; ++ni) bf[ni] = frag_ld(Bt, wn + ni * 16, lane);
    for (int mi = 0; mi < 4; ++mi)
      for (int ni = 0; ni < 4; ++ni)
        acc[mi][ni] = __builtin_amdgcn_wmma_f32_16x16x32_f16(
            false, af[mi], false, bf[ni], (short)0, acc[mi][ni], false, false);
    __syncthreads();
  }

  _Float16* dstb = dst + (size_t)b * CCH * HW;
  for (int mi = 0; mi < 4; ++mi)
    for (int ni = 0; ni < 4; ++ni) {
      int col  = n0 + wn + ni * 16 + (lane & 15);
      int rowB = m0 + wm + mi * 16 + 8 * (lane >> 4);
      for (int r = 0; r < 8; ++r) {
        int row = rowB + r;
        float val = acc[mi][ni][r] + bias[row];
        dstb[(size_t)row * HW + col] = (_Float16)val;
      }
    }
}

// ===========================================================================
// Stage 2: S = Q * K^T  (M=N=512, K=1024), f16 in -> f32 logits.
// Both operands K-contiguous: pure 16B copies -> async global->LDS path.
// Block tile 128x256. grid = (8 tiles, 64 batch), block = 256.
// ===========================================================================
__global__ __launch_bounds__(256)
void gemm_qkt(const _Float16* __restrict__ Q, const _Float16* __restrict__ Kd,
              float* __restrict__ S) {
  __shared__ _Float16 At[128 * LDSS];
  __shared__ _Float16 Bt[256 * LDSS];

  const int tid  = threadIdx.x;
  const int lane = tid & 31;
  const int wave = tid >> 5;
  const int wm   = (wave & 1) * 64;
  const int wn   = (wave >> 1) * 64;
  const int m0   = (blockIdx.x & 3) * 128;
  const int n0   = (blockIdx.x >> 2) * 256;
  const int b    = blockIdx.y;

  const _Float16* Qb = Q  + (size_t)b * CCH * HW;
  const _Float16* Kb = Kd + (size_t)b * CCH * HW;

  v8f acc[4][4];
  for (int i = 0; i < 4; ++i)
    for (int j = 0; j < 4; ++j) acc[i][j] = (v8f)(0.0f);

  for (int kt = 0; kt < HW; kt += 32) {
    for (int i = 0; i < 2; ++i) {               // A: 512 16B slots
      int slot = tid + i * 256;
      int r    = slot >> 2;
      int c8   = (slot & 3) << 3;
      cp16_g2l(Qb + (size_t)(m0 + r) * HW + kt + c8, &At[r * LDSS + c8]);
    }
    for (int i = 0; i < 4; ++i) {               // B: 1024 16B slots
      int slot = tid + i * 256;
      int r    = slot >> 2;
      int c8   = (slot & 3) << 3;
      cp16_g2l(Kb + (size_t)(n0 + r) * HW + kt + c8, &Bt[r * LDSS + c8]);
    }
    cp16_wait();
    __syncthreads();

    v16h af[4], bf[4];
    for (int mi = 0; mi < 4; ++mi) af[mi] = frag_ld(At, wm + mi * 16, lane);
    for (int ni = 0; ni < 4; ++ni) bf[ni] = frag_ld(Bt, wn + ni * 16, lane);
    for (int mi = 0; mi < 4; ++mi)
      for (int ni = 0; ni < 4; ++ni)
        acc[mi][ni] = __builtin_amdgcn_wmma_f32_16x16x32_f16(
            false, af[mi], false, bf[ni], (short)0, acc[mi][ni], false, false);
    __syncthreads();
  }

  float* Sb = S + (size_t)b * CCH * CCH;
  for (int mi = 0; mi < 4; ++mi)
    for (int ni = 0; ni < 4; ++ni) {
      int col  = n0 + wn + ni * 16 + (lane & 15);
      int rowB = m0 + wm + mi * 16 + 8 * (lane >> 4);
      for (int r = 0; r < 8; ++r)
        Sb[(size_t)(rowB + r) * CCH + col] = acc[mi][ni][r];
    }
}

// ===========================================================================
// Stage 3: P = softmax_rows(S), f32 -> f16. One wave per 512-wide row.
// ===========================================================================
__global__ __launch_bounds__(256)
void softmax_rows(const float* __restrict__ S, _Float16* __restrict__ P) {
  const int lane = threadIdx.x & 31;
  const int row  = blockIdx.x * 8 + (threadIdx.x >> 5);
  const float* s = S + (size_t)row * CCH;

  float v[16];
  float m = -1e30f;
  for (int j = 0; j < 16; ++j) { v[j] = s[lane + 32 * j]; m = fmaxf(m, v[j]); }
  for (int off = 16; off > 0; off >>= 1) m = fmaxf(m, __shfl_xor(m, off, 32));
  float sum = 0.0f;
  for (int j = 0; j < 16; ++j) { v[j] = __expf(v[j] - m); sum += v[j]; }
  for (int off = 16; off > 0; off >>= 1) sum += __shfl_xor(sum, off, 32);
  float inv = 1.0f / sum;
  _Float16* p = P + (size_t)row * CCH;
  for (int j = 0; j < 16; ++j) p[lane + 32 * j] = (_Float16)(v[j] * inv);
}

// ===========================================================================
// Stage 4: out = gamma * (P * V) + x   (M=512, N=1024, K=512).
// A = P (K-contiguous f16, async copies), B = V (transpose on fill).
// Block tile 128x256. grid = (16 tiles, 64 batch), block = 256.
// ===========================================================================
__global__ __launch_bounds__(256)
void gemm_pv(const _Float16* __restrict__ P, const _Float16* __restrict__ V,
             const float* __restrict__ img, const float* __restrict__ gamma,
             float* __restrict__ out) {
  __shared__ _Float16 At[128 * LDSS];
  __shared__ _Float16 Bt[256 * LDSS];

  const int tid  = threadIdx.x;
  const int lane = tid & 31;
  const int wave = tid >> 5;
  const int wm   = (wave & 1) * 64;
  const int wn   = (wave >> 1) * 64;
  const int m0   = (blockIdx.x & 3) * 128;
  const int n0   = (blockIdx.x >> 2) * 256;
  const int b    = blockIdx.y;

  const _Float16* Pb = P + (size_t)b * CCH * CCH;
  const _Float16* Vb = V + (size_t)b * CCH * HW;

  v8f acc[4][4];
  for (int i = 0; i < 4; ++i)
    for (int j = 0; j < 4; ++j) acc[i][j] = (v8f)(0.0f);

  for (int kt = 0; kt < CCH; kt += 32) {
    // A: P rows [m0,m0+128), K-contiguous -> async copies
    for (int i = 0; i < 2; ++i) {
      int slot = tid + i * 256;
      int r    = slot >> 2;
      int c8   = (slot & 3) << 3;
      cp16_g2l(Pb + (size_t)(m0 + r) * CCH + kt + c8, &At[r * LDSS + c8]);
    }
    // B: V rows [kt,kt+32) x cols [n0,n0+256), transpose into [n][k]
    for (int i = 0; i < 4; ++i) {
      int slot = tid + i * 256;                 // 32 k-rows * 32 chunks of 8
      int kr   = slot >> 5;
      int n8   = (slot & 31) << 3;
      v8h v = *(const v8h*)(Vb + (size_t)(kt + kr) * HW + n0 + n8);
      for (int j = 0; j < 8; ++j) Bt[(n8 + j) * LDSS + kr] = v[j];
    }
    cp16_wait();
    __syncthreads();

    v16h af[4], bf[4];
    for (int mi = 0; mi < 4; ++mi) af[mi] = frag_ld(At, wm + mi * 16, lane);
    for (int ni = 0; ni < 4; ++ni) bf[ni] = frag_ld(Bt, wn + ni * 16, lane);
    for (int mi = 0; mi < 4; ++mi)
      for (int ni = 0; ni < 4; ++ni)
        acc[mi][ni] = __builtin_amdgcn_wmma_f32_16x16x32_f16(
            false, af[mi], false, bf[ni], (short)0, acc[mi][ni], false, false);
    __syncthreads();
  }

  const float g = gamma[0];
  const float* xb = img + (size_t)b * CCH * HW;
  float* ob = out + (size_t)b * CCH * HW;
  for (int mi = 0; mi < 4; ++mi)
    for (int ni = 0; ni < 4; ++ni) {
      int col  = n0 + wn + ni * 16 + (lane & 15);
      int rowB = m0 + wm + mi * 16 + 8 * (lane >> 4);
      for (int r = 0; r < 8; ++r) {
        size_t idx = (size_t)(rowB + r) * HW + col;
        ob[idx] = g * acc[mi][ni][r] + xb[idx];
      }
    }
}

// ===========================================================================
extern "C" void kernel_launch(void* const* d_in, const int* in_sizes, int n_in,
                              void* d_out, int out_size, void* d_ws, size_t ws_size,
                              hipStream_t stream) {
  const float* img   = (const float*)d_in[0];
  const float* dep   = (const float*)d_in[1];
  const float* Wq    = (const float*)d_in[2];
  const float* bq    = (const float*)d_in[3];
  const float* Wk    = (const float*)d_in[4];
  const float* bk    = (const float*)d_in[5];
  const float* Wv    = (const float*)d_in[6];
  const float* bv    = (const float*)d_in[7];
  const float* gamma = (const float*)d_in[8];

  const size_t QKV_BYTES = (size_t)NB * CCH * HW * sizeof(_Float16);  // 64 MB each
  char* ws = (char*)d_ws;
  _Float16* Q  = (_Float16*)(ws);
  _Float16* Kd = (_Float16*)(ws + QKV_BYTES);
  _Float16* V  = (_Float16*)(ws + 2 * QKV_BYTES);
  float*    S  = (float*)   (ws + 3 * QKV_BYTES);                     // 64 MB
  _Float16* P  = Q;   // Q is dead after stage 2; reuse its region for P

  gemm_qkv<<<dim3(16, 3, NB), 256, 0, stream>>>(Wq, bq, Wk, bk, Wv, bv,
                                                img, dep, Q, Kd, V);
  gemm_qkt<<<dim3(8, NB), 256, 0, stream>>>(Q, Kd, S);
  softmax_rows<<<dim3(NB * CCH / 8), 256, 0, stream>>>(S, P);
  gemm_pv<<<dim3(16, NB), 256, 0, stream>>>(P, V, img, gamma, (float*)d_out);
}
